// InvariantPointAttention_73718818668718
// MI455X (gfx1250) — compile-verified, matrix operand order checked by reading, not run on hardware
//
#include <hip/hip_runtime.h>
#include <hip/hip_bf16.h>
#include <math.h>

#define B_  2
#define N_  512
#define CS  384
#define CZ  128
#define H_  12
#define D_  16
#define PQ_ 4
#define PV_ 8
#define FEAT_DIM 2112   // H*D + 4*H*PV + H*CZ
#define VF  48          // padded [v(16) | vg(24) | 0(8)] columns

typedef __attribute__((ext_vector_type(16))) __bf16 bf16x16;
typedef __attribute__((ext_vector_type(8)))  __bf16 bf16x8;
typedef __attribute__((ext_vector_type(4)))  __bf16 bf16x4;
typedef __attribute__((ext_vector_type(2)))  __bf16 bf16x2;
typedef __attribute__((ext_vector_type(8)))  float  floatx8;

// ---------------------------------------------------------------------------
// WMMA: D = A(16x32 bf16) * B(32x16 bf16) + C(16x16 f32)
// ---------------------------------------------------------------------------
__device__ __forceinline__ floatx8 wmma_bf16(bf16x16 a, bf16x16 b, floatx8 c) {
  return __builtin_amdgcn_wmma_f32_16x16x32_bf16(
      false, a, false, b, (short)0, c, false, false);
}

// Fragment from a row-major 16x32 bf16 tile (A: rows=M, B: rows=N), any stride.
// ISA 7.12.2: lane half h covers K = h*8+0..7 (elems 0-7) and 16+h*8+0..7
// (elems 8-15) -> two contiguous 16B loads (b128).
__device__ __forceinline__ bf16x16 frag_rm(const __bf16* T, size_t stride) {
  const int lane = threadIdx.x & 31;
  const int row = lane & 15, half = lane >> 4;
  const __bf16* p = T + (size_t)row * stride + half * 8;
  const bf16x8 lo = *(const bf16x8*)(p);
  const bf16x8 hi = *(const bf16x8*)(p + 16);
  bf16x16 f;
#pragma unroll
  for (int e = 0; e < 8; ++e) { f[e] = lo[e]; f[e + 8] = hi[e]; }
  return f;
}

// Branch-free stage: 16x32 fp32 tile (row stride ld) -> LDS bf16 16x32.
__device__ __forceinline__ void stage_rm(const float* __restrict__ src, int ld,
                                         __bf16* __restrict__ dst) {
  const int lane = threadIdx.x & 31;
#pragma unroll
  for (int i = 0; i < 4; ++i) {
    const int q = lane + i * 32;       // 0..127 float4 chunks
    const int r = q >> 3;              // row 0..15
    const int c = (q & 7) * 4;         // col 0..28
    const float4 v = *(const float4*)(src + (size_t)r * ld + c);
    bf16x4 pk;
    pk[0] = (__bf16)v.x; pk[1] = (__bf16)v.y; pk[2] = (__bf16)v.z; pk[3] = (__bf16)v.w;
    *(bf16x4*)(dst + r * 32 + c) = pk;
  }
}

// Branch-free transposing stage: 32(K)x16(N) fp32 (row stride ld) -> LDS bf16
// 16(N)x32(K) row-major. Coalesced float4 reads.
__device__ __forceinline__ void stage_tr(const float* __restrict__ src, int ld,
                                         __bf16* __restrict__ dst) {
  const int lane = threadIdx.x & 31;
#pragma unroll
  for (int i = 0; i < 4; ++i) {
    const int q = lane + i * 32;
    const int k = q >> 2;              // 0..31
    const int n = (q & 3) * 4;         // 0,4,8,12
    const float4 v = *(const float4*)(src + (size_t)k * ld + n);
    dst[(n + 0) * 32 + k] = (__bf16)v.x;
    dst[(n + 1) * 32 + k] = (__bf16)v.y;
    dst[(n + 2) * 32 + k] = (__bf16)v.z;
    dst[(n + 3) * 32 + k] = (__bf16)v.w;
  }
}

// Guarded transpose stage (uniform scalar branch only) for partial-N tiles.
__device__ __forceinline__ void stage_tr_g(const float* __restrict__ src, int ld,
                                           int nvalid, __bf16* __restrict__ dst) {
  const int k = threadIdx.x & 31;
#pragma unroll
  for (int n = 0; n < 16; ++n) {
    float v = 0.f;
    if (n < nvalid) v = src[(size_t)k * ld + n];
    dst[n * 32 + k] = (__bf16)v;
  }
}

// ---------------------------------------------------------------------------
// Generic GEMM: C[MxNc] = A[MxK] @ W[KxNc] + bias ; optional row mask.
// K must be a multiple of 32 (384, 128, 2112 at call sites).
// ---------------------------------------------------------------------------
__global__ void gemm_kernel(const float* __restrict__ A, const float* __restrict__ W,
                            const float* __restrict__ bias, const float* __restrict__ rowmask,
                            float* __restrict__ C, int M, int K, int Nc) {
  __shared__ __bf16 As[16 * 32];
  __shared__ __bf16 Bs[16 * 32];       // n-major (16 rows of K=32)
  const int lane = threadIdx.x;
  const int m0 = blockIdx.x * 16;
  const int n0 = blockIdx.y * 16;
  const bool fullN = (n0 + 16 <= Nc);
  floatx8 acc = {};
  for (int k0 = 0; k0 < K; k0 += 32) {
    stage_rm(A + (size_t)m0 * K + k0, K, As);
    if (fullN) stage_tr(W + (size_t)k0 * Nc + n0, Nc, Bs);
    else       stage_tr_g(W + (size_t)k0 * Nc + n0, Nc, Nc - n0, Bs);
    __syncthreads();
    acc = wmma_bf16(frag_rm(As, 32), frag_rm(Bs, 32), acc);
    __syncthreads();
  }
  const int n = lane & 15, half = lane >> 4;
  if (n0 + n < Nc) {
    const float bv = bias ? bias[n0 + n] : 0.f;
#pragma unroll
    for (int r = 0; r < 8; ++r) {
      const int m = m0 + half * 8 + r;
      float res = acc[r] + bv;
      if (rowmask) res *= rowmask[m];
      C[(size_t)m * Nc + n0 + n] = res;
    }
  }
}

// ---------------------------------------------------------------------------
// Local -> global point frames: g = R @ p + t
// ---------------------------------------------------------------------------
__global__ void points_kernel(const float* __restrict__ qpraw, const float* __restrict__ kvpraw,
                              const float* __restrict__ R, const float* __restrict__ t,
                              float* __restrict__ qg, float* __restrict__ kg,
                              float* __restrict__ vg) {
  const int tid = blockIdx.x * blockDim.x + threadIdx.x;
  const int total = B_ * N_ * H_ * 16;
  if (tid >= total) return;
  const int slot = tid & 15;
  const int h = (tid >> 4) % H_;
  const int bn = tid / (16 * H_);
  float lx, ly, lz;
  if (slot < PQ_) {
    const float* base = qpraw + (size_t)bn * (3 * H_ * PQ_);
    lx = base[0 * H_ * PQ_ + h * PQ_ + slot];
    ly = base[1 * H_ * PQ_ + h * PQ_ + slot];
    lz = base[2 * H_ * PQ_ + h * PQ_ + slot];
  } else {
    const int pp = slot - PQ_;
    const float* base = kvpraw + (size_t)bn * (3 * H_ * 12);
    lx = base[0 * H_ * 12 + h * 12 + pp];
    ly = base[1 * H_ * 12 + h * 12 + pp];
    lz = base[2 * H_ * 12 + h * 12 + pp];
  }
  const float* Rb = R + (size_t)bn * 9;
  const float* tb = t + (size_t)bn * 3;
  const float gx = Rb[0] * lx + Rb[1] * ly + Rb[2] * lz + tb[0];
  const float gy = Rb[3] * lx + Rb[4] * ly + Rb[5] * lz + tb[1];
  const float gz = Rb[6] * lx + Rb[7] * ly + Rb[8] * lz + tb[2];
  float* o;
  if (slot < PQ_)            o = qg + (((size_t)bn * H_ + h) * PQ_ + slot) * 3;
  else if (slot < 2 * PQ_)   o = kg + (((size_t)bn * H_ + h) * PQ_ + (slot - PQ_)) * 3;
  else                       o = vg + (((size_t)bn * H_ + h) * PV_ + (slot - 2 * PQ_)) * 3;
  o[0] = gx; o[1] = gy; o[2] = gz;
}

// ---------------------------------------------------------------------------
// 32-wide logit features (WMMA-ready row-major n x 32):
// a = [q*s1, hw*qg(12), -0.5*hw*|qg|^2, 1, 0, 0]
// b = [k,    kg(12),    1, -0.5*hw*|kg|^2, 0, 0]
// ---------------------------------------------------------------------------
__global__ void feat_kernel(const float* __restrict__ qraw, const float* __restrict__ kvraw,
                            const float* __restrict__ qg, const float* __restrict__ kg,
                            const float* __restrict__ hwts,
                            __bf16* __restrict__ afeat, __bf16* __restrict__ bfeat) {
  const int tid = blockIdx.x * blockDim.x + threadIdx.x;
  const int total = B_ * H_ * N_;
  if (tid >= total) return;
  const int n = tid % N_;
  const int h = (tid / N_) % H_;
  const int b = tid / (N_ * H_);
  const float x = hwts[h];
  const float sp = (x > 20.f) ? x : log1pf(expf(x));
  const float hw = sp * 0.13608276348795434f;   // sqrt(1/54)
  const float s1 = 0.14433756729740643f;        // sqrt(1/48)
  const int row = b * N_ + n;
  const float* q   = qraw  + (size_t)row * 192 + h * 16;
  const float* kk  = kvraw + (size_t)row * 384 + h * 32;
  const float* qgv = qg + ((size_t)row * H_ + h) * PQ_ * 3;
  const float* kgv = kg + ((size_t)row * H_ + h) * PQ_ * 3;
  float qn2 = 0.f, kn2 = 0.f;
#pragma unroll
  for (int e = 0; e < 12; ++e) { qn2 += qgv[e] * qgv[e]; kn2 += kgv[e] * kgv[e]; }
  __bf16* a  = afeat + (size_t)tid * 32;
  __bf16* bo = bfeat + (size_t)tid * 32;
#pragma unroll
  for (int d = 0; d < 16; ++d) { a[d] = (__bf16)(q[d] * s1); bo[d] = (__bf16)kk[d]; }
#pragma unroll
  for (int e = 0; e < 12; ++e) { a[16 + e] = (__bf16)(qgv[e] * hw); bo[16 + e] = (__bf16)kgv[e]; }
  a[28] = (__bf16)(-0.5f * hw * qn2);  a[29] = (__bf16)1.0f;
  a[30] = (__bf16)0.0f;                a[31] = (__bf16)0.0f;
  bo[28] = (__bf16)1.0f;               bo[29] = (__bf16)(-0.5f * hw * kn2);
  bo[30] = (__bf16)0.0f;               bo[31] = (__bf16)0.0f;
}

// ---------------------------------------------------------------------------
// Pack [v | vg | 0] column-major: vfT[(bh*48 + col)*512 + j]  (bf16)
// ---------------------------------------------------------------------------
__global__ void vfeat_kernel(const float* __restrict__ kvraw, const float* __restrict__ vg,
                             __bf16* __restrict__ vfT) {
  const int tid = blockIdx.x * blockDim.x + threadIdx.x;
  const int total = B_ * H_ * VF * N_;
  if (tid >= total) return;
  const int j   = tid % N_;
  const int col = (tid / N_) % VF;
  const int bh  = tid / (N_ * VF);
  const int b = bh / H_, h = bh % H_;
  float v = 0.f;
  if (col < D_)                v = kvraw[(size_t)(b * N_ + j) * 384 + h * 32 + 16 + col];
  else if (col < D_ + PV_ * 3) v = vg[((size_t)(b * N_ + j) * H_ + h) * PV_ * 3 + (col - D_)];
  vfT[tid] = (__bf16)v;
}

// ---------------------------------------------------------------------------
// Logits (one K=32 WMMA per tile, operands direct from global) +
// pair-bias + mask + row softmax -> bf16 attention.
// ---------------------------------------------------------------------------
__global__ void attn_kernel(const __bf16* __restrict__ afeat, const __bf16* __restrict__ bfeat,
                            const float* __restrict__ biasbuf, const float* __restrict__ mask,
                            __bf16* __restrict__ attnbf) {
  __shared__ float L[16 * 512];
  const int lane = threadIdx.x;
  const int i0 = blockIdx.x * 16;
  const int h = blockIdx.y;
  const int b = blockIdx.z;
  const int bh = b * H_ + h;
  const bf16x16 fa = frag_rm(afeat + ((size_t)bh * N_ + i0) * 32, 32);
  const int n = lane & 15, half = lane >> 4;
  for (int jt = 0; jt < N_ / 16; ++jt) {
    const int j0 = jt * 16;
    const bf16x16 fb = frag_rm(bfeat + ((size_t)bh * N_ + j0) * 32, 32);
    floatx8 acc = {};
    acc = wmma_bf16(fa, fb, acc);
    const int j = j0 + n;
    const float mj = mask[b * N_ + j];
#pragma unroll
    for (int r = 0; r < 8; ++r) {
      const int m = half * 8 + r;
      const int i = i0 + m;
      const float mrow = mask[b * N_ + i];
      const float l = acc[r]
                    + 0.5773502691896258f * biasbuf[((size_t)(b * N_ + i) * N_ + j) * H_ + h]
                    + 100000.0f * (mrow * mj - 1.0f);
      L[m * 512 + j] = l;
    }
  }
  __syncthreads();
  if (lane < 16) {
    float* row = &L[lane * 512];
    float mx = -3.0e38f;
    for (int j = 0; j < 512; ++j) mx = fmaxf(mx, row[j]);
    float s = 0.f;
    for (int j = 0; j < 512; ++j) { const float e = expf(row[j] - mx); row[j] = e; s += e; }
    const float inv = 1.0f / s;
    __bf16* o = attnbf + ((size_t)bh * N_ + i0 + lane) * N_;
    for (int j = 0; j < 512; j += 2) {
      bf16x2 pk;
      pk[0] = (__bf16)(row[j] * inv);
      pk[1] = (__bf16)(row[j + 1] * inv);
      *(bf16x2*)(o + j) = pk;
    }
  }
}

// ---------------------------------------------------------------------------
// out_scalar + rpg: attn(512x512) @ [v|vg|0](512x48), per (b,h).
// Pure global-fragment WMMA loop: no LDS, no barriers.
// ---------------------------------------------------------------------------
__global__ void av_kernel(const __bf16* __restrict__ attnbf, const __bf16* __restrict__ vfT,
                          float* __restrict__ feats, float* __restrict__ rpg) {
  const int lane = threadIdx.x;
  const int i0 = blockIdx.x * 16;
  const int nt = blockIdx.y;            // 0..2
  const int bh = blockIdx.z;
  const int b = bh / H_, h = bh % H_;
  const __bf16* Abase = attnbf + ((size_t)bh * N_ + i0) * N_;
  const __bf16* Bbase = vfT + ((size_t)bh * VF + nt * 16) * N_;
  floatx8 acc = {};
  for (int j0 = 0; j0 < N_; j0 += 32) {
    acc = wmma_bf16(frag_rm(Abase + j0, N_), frag_rm(Bbase + j0, N_), acc);
  }
  const int n = lane & 15, half = lane >> 4;
  const int col = nt * 16 + n;
#pragma unroll
  for (int r = 0; r < 8; ++r) {
    const int i = i0 + half * 8 + r;
    if (col < D_)
      feats[(size_t)(b * N_ + i) * FEAT_DIM + h * D_ + col] = acc[r];
    else if (col < D_ + PV_ * 3)
      rpg[((size_t)(b * N_ + i) * H_ + h) * PV_ * 3 + (col - D_)] = acc[r];
  }
}

// ---------------------------------------------------------------------------
// rpl = R^T (rpg - t); x/y/z slices + norms into feats.
// ---------------------------------------------------------------------------
__global__ void rpl_kernel(const float* __restrict__ rpg, const float* __restrict__ R,
                           const float* __restrict__ t, float* __restrict__ feats) {
  const int tid = blockIdx.x * blockDim.x + threadIdx.x;
  const int total = B_ * N_ * H_ * PV_;
  if (tid >= total) return;
  const int p = tid % PV_;
  const int h = (tid / PV_) % H_;
  const int bn = tid / (PV_ * H_);
  const float* g = rpg + (((size_t)bn * H_ + h) * PV_ + p) * 3;
  const float* Rb = R + (size_t)bn * 9;
  const float* tb = t + (size_t)bn * 3;
  const float dx = g[0] - tb[0], dy = g[1] - tb[1], dz = g[2] - tb[2];
  const float lx = Rb[0] * dx + Rb[3] * dy + Rb[6] * dz;
  const float ly = Rb[1] * dx + Rb[4] * dy + Rb[7] * dz;
  const float lz = Rb[2] * dx + Rb[5] * dy + Rb[8] * dz;
  const float nm = sqrtf(lx * lx + ly * ly + lz * lz + 1e-8f);
  float* f = feats + (size_t)bn * FEAT_DIM;
  const int hp = h * PV_ + p;
  f[192 + hp] = lx; f[288 + hp] = ly; f[384 + hp] = lz; f[480 + hp] = nm;
}

// ---------------------------------------------------------------------------
// Issue 4 async b128 copies: one 32(j)x16(c) fp32 z tile -> LDS (row-major,
// chunk q at LDS byte offset q*16). Tracked on ASYNCcnt.
// ---------------------------------------------------------------------------
__device__ __forceinline__ void async_ztile(const float* gbase, unsigned ldsbase) {
  const int lane = threadIdx.x & 31;
#pragma unroll
  for (int i = 0; i < 4; ++i) {
    const int q = lane + i * 32;       // 0..127
    const int k = q >> 2;              // j row 0..31
    const int n4 = (q & 3) * 4;        // c chunk
    const float* g = gbase + (size_t)k * CZ + n4;
    const unsigned l = ldsbase + (unsigned)q * 16;
    asm volatile("global_load_async_to_lds_b128 %0, %1, off"
                 :: "v"(l), "v"(g) : "memory");
  }
}

// ---------------------------------------------------------------------------
// out_pair: attn[h(16 pad), j] (16x512) @ z[b,i,j,:] (512x128), per (b,i).
// z streamed with double-buffered GLOBAL_LOAD_ASYNC_TO_LDS_B128 (ASYNCcnt),
// converted/transposed LDS->LDS, consumed as b128 fragments.
// ---------------------------------------------------------------------------
__global__ void op_kernel(const __bf16* __restrict__ attnbf, const float* __restrict__ z,
                          float* __restrict__ feats) {
  __shared__ float  Zs[2][32 * 16];    // raw fp32 tiles (async landing buffers)
  __shared__ __bf16 Bs[16 * 32];       // transposed bf16 tile
  const int lane = threadIdx.x;
  const int c0 = blockIdx.x * 16;
  const int i = blockIdx.y;
  const int b = blockIdx.z;
  const size_t zrow = ((size_t)b * N_ + i) * N_;
  const float* zbase = z + zrow * CZ + c0;
  const __bf16* Abase = attnbf + ((size_t)(b * H_) * N_ + i) * N_;  // row stride N*N
  const unsigned lds0 = (unsigned)(uintptr_t)&Zs[0][0];
  const unsigned lds1 = (unsigned)(uintptr_t)&Zs[1][0];
  async_ztile(zbase, lds0);            // prologue: tile 0 -> buf 0
  floatx8 acc = {};
  for (int jt = 0; jt < N_ / 32; ++jt) {
    const int j0 = jt * 32;
    if (jt + 1 < N_ / 32) {
      async_ztile(zbase + (size_t)(j0 + 32) * CZ, (jt & 1) ? lds0 : lds1);
      asm volatile("s_wait_asynccnt 0x4" ::: "memory");   // tile jt resident
    } else {
      asm volatile("s_wait_asynccnt 0x0" ::: "memory");
    }
    const float* Zt = Zs[jt & 1];
#pragma unroll
    for (int ii = 0; ii < 4; ++ii) {   // convert + transpose LDS->LDS
      const int q = lane + ii * 32;
      const int k = q >> 2, n4 = (q & 3) * 4;
      const float4 v = *(const float4*)(Zt + q * 4);
      Bs[(n4 + 0) * 32 + k] = (__bf16)v.x;
      Bs[(n4 + 1) * 32 + k] = (__bf16)v.y;
      Bs[(n4 + 2) * 32 + k] = (__bf16)v.z;
      Bs[(n4 + 3) * 32 + k] = (__bf16)v.w;
    }
    acc = wmma_bf16(frag_rm(Abase + j0, (size_t)N_ * N_), frag_rm(Bs, 32), acc);
  }
  const int n = lane & 15, half = lane >> 4;
#pragma unroll
  for (int r = 0; r < 8; ++r) {
    const int hh = half * 8 + r;
    if (hh < H_)
      feats[((size_t)b * N_ + i) * FEAT_DIM + 576 + hh * CZ + c0 + n] = acc[r];
  }
}

// ---------------------------------------------------------------------------
extern "C" void kernel_launch(void* const* d_in, const int* in_sizes, int n_in,
                              void* d_out, int out_size, void* d_ws, size_t ws_size,
                              hipStream_t stream) {
  const float* s    = (const float*)d_in[0];
  const float* z    = (const float*)d_in[1];
  const float* R    = (const float*)d_in[2];
  const float* t    = (const float*)d_in[3];
  const float* mask = (const float*)d_in[4];
  const float* wq   = (const float*)d_in[5];
  const float* bq   = (const float*)d_in[6];
  const float* wkv  = (const float*)d_in[7];
  const float* bkv  = (const float*)d_in[8];
  const float* wqp  = (const float*)d_in[9];
  const float* bqp  = (const float*)d_in[10];
  const float* wkvp = (const float*)d_in[11];
  const float* bkvp = (const float*)d_in[12];
  const float* wb   = (const float*)d_in[13];
  const float* bb   = (const float*)d_in[14];
  const float* wout = (const float*)d_in[15];
  const float* bout = (const float*)d_in[16];
  const float* hwts = (const float*)d_in[17];
  float* out = (float*)d_out;

  char* wp = (char*)d_ws;
  auto alloc = [&](size_t bytes) {
    char* p = wp;
    wp += (bytes + 255) & ~(size_t)255;
    return p;
  };
  const int M = B_ * N_;  // 1024
  float*  qraw    = (float*)alloc((size_t)M * 192 * 4);
  float*  kvraw   = (float*)alloc((size_t)M * 384 * 4);
  float*  qpraw   = (float*)alloc((size_t)M * 144 * 4);
  float*  kvpraw  = (float*)alloc((size_t)M * 432 * 4);
  float*  qg      = (float*)alloc((size_t)M * H_ * PQ_ * 3 * 4);
  float*  kg      = (float*)alloc((size_t)M * H_ * PQ_ * 3 * 4);
  float*  vg      = (float*)alloc((size_t)M * H_ * PV_ * 3 * 4);
  float*  biasbuf = (float*)alloc((size_t)B_ * N_ * N_ * H_ * 4);
  __bf16* attnbf  = (__bf16*)alloc((size_t)(B_ * H_ + 4) * N_ * N_ * 2);  // +4 pad planes
  float*  rpg     = (float*)alloc((size_t)M * H_ * PV_ * 3 * 4);
  float*  feats   = (float*)alloc((size_t)M * FEAT_DIM * 4);
  __bf16* afeat   = (__bf16*)alloc((size_t)B_ * H_ * N_ * 32 * 2);
  __bf16* bfeat   = (__bf16*)alloc((size_t)B_ * H_ * N_ * 32 * 2);
  __bf16* vfT     = (__bf16*)alloc((size_t)B_ * H_ * VF * N_ * 2);

  const dim3 wv(32, 1, 1);

  // 1-4: projections s @ {wq,wkv,wqp,wkvp}
  gemm_kernel<<<dim3(M / 16, 192 / 16), wv, 0, stream>>>(s, wq,   bq,   nullptr, qraw,   M, CS, 192);
  gemm_kernel<<<dim3(M / 16, 384 / 16), wv, 0, stream>>>(s, wkv,  bkv,  nullptr, kvraw,  M, CS, 384);
  gemm_kernel<<<dim3(M / 16, 144 / 16), wv, 0, stream>>>(s, wqp,  bqp,  nullptr, qpraw,  M, CS, 144);
  gemm_kernel<<<dim3(M / 16, 432 / 16), wv, 0, stream>>>(s, wkvp, bkvp, nullptr, kvpraw, M, CS, 432);

  // 5: point frames
  {
    const int total = M * H_ * 16;
    points_kernel<<<(total + 255) / 256, 256, 0, stream>>>(qpraw, kvpraw, R, t, qg, kg, vg);
  }
  // 6: logit features
  {
    const int total = B_ * H_ * N_;
    feat_kernel<<<(total + 255) / 256, 256, 0, stream>>>(qraw, kvraw, qg, kg, hwts, afeat, bfeat);
  }
  // 6b: packed V features (column-major bf16)
  {
    const int total = B_ * H_ * VF * N_;
    vfeat_kernel<<<(total + 255) / 256, 256, 0, stream>>>(kvraw, vg, vfT);
  }
  // 7: pair bias = z @ wb + bb
  gemm_kernel<<<dim3((B_ * N_ * N_) / 16, 1), wv, 0, stream>>>(z, wb, bb, nullptr, biasbuf,
                                                               B_ * N_ * N_, CZ, H_);
  // 8: logits + softmax -> bf16 attn
  attn_kernel<<<dim3(N_ / 16, H_, B_), wv, 0, stream>>>(afeat, bfeat, biasbuf, mask, attnbf);
  // 9: attn @ [v|vg]
  av_kernel<<<dim3(N_ / 16, 3, B_ * H_), wv, 0, stream>>>(attnbf, vfT, feats, rpg);
  // 10: inverse frame + norms
  {
    const int total = M * H_ * PV_;
    rpl_kernel<<<(total + 255) / 256, 256, 0, stream>>>(rpg, R, t, feats);
  }
  // 11: out_pair = attn @ z (async double-buffered z stream)
  op_kernel<<<dim3(CZ / 16, N_, B_), wv, 0, stream>>>(attnbf, z, feats);
  // 12: out = (feats @ wout + bout) * mask
  gemm_kernel<<<dim3(M / 16, 384 / 16), wv, 0, stream>>>(feats, wout, bout, mask, out,
                                                         M, FEAT_DIM, CS);
}
